// FullLanguageZone_54485955117491
// MI455X (gfx1250) — matrix operand order repaired and checked by reading, not compiled, not run on hardware
//
#include <hip/hip_runtime.h>
#include <hip/hip_bf16.h>
#include <math.h>
#include <stdint.h>

typedef __bf16 bf16_t;
typedef __bf16 v16bf __attribute__((ext_vector_type(16)));
typedef float  v8f   __attribute__((ext_vector_type(8)));

// Sizes (fixed by the reference)
static constexpr int NB = 4, SSEQ = 2048, DDIM = 1024, HDIM = 2048;
static constexpr int NTOK = NB * SSEQ;      // 8192
static constexpr int NEXP = 8, MMOE = 64, HHALF = 1024;

// ---------------------------------------------------------------- gains
__global__ void k_gains(const int* __restrict__ ids, const float* __restrict__ tab,
                        float* __restrict__ gains, int n) {
    int i = blockIdx.x * blockDim.x + threadIdx.x;
    if (i < n) gains[i] = 1.0f + tanhf(tab[ids[i]]);
}

// ------------------------------------------------ transpose f32 -> bf16 (Wt[n][k] = W[k][n])
__global__ void k_transpose_bf16(const float* __restrict__ in, bf16_t* __restrict__ out,
                                 int batch, int rows, int cols) {
    size_t rc = (size_t)rows * cols;
    size_t total = (size_t)batch * rc;
    for (size_t i = (size_t)blockIdx.x * blockDim.x + threadIdx.x; i < total;
         i += (size_t)gridDim.x * blockDim.x) {
        size_t bt = i / rc, rem = i - bt * rc;
        size_t c = rem / rows, r = rem - c * rows;       // output [bt][c][r]
        out[i] = (bf16_t)in[bt * rc + r * cols + c];
    }
}

// ---------------------------------------------------------------- generic WMMA GEMM
// 32x32 output block per wave: 2 M-tiles x 2 N-tiles -> 4 accumulators, so each
// 32-wide k-step issues 4 v_wmma for 2 A-frag + 2 B-frag loads (2x reuse vs 16x16).
// AMODE: 0 = A f32, 1 = A f32 scaled by gains[row], 2 = A bf16
// EPI:   0 = store f32,  1 = store bf16( sigmoid(acc) * gains[row] )
// C initialized with bias[col]. Bt is column-major weights: Bt[n*K + k].
template <int AMODE, int EPI>
__global__ void k_gemm_wmma(const void* __restrict__ Av, const bf16_t* __restrict__ Bt,
                            const float* __restrict__ bias, const float* __restrict__ gains,
                            void* __restrict__ Out, int Mrows, int N, int K) {
    const int lane = threadIdx.x & 31;
    const int wave = (int)(((size_t)blockIdx.x * blockDim.x + threadIdx.x) >> 5);
    const int ntiles2 = N >> 5;                 // 32-col blocks
    const int tm = wave / ntiles2, tn = wave - tm * ntiles2;
    const int m0 = tm * 32;
    if (m0 >= Mrows) return;
    const int n0 = tn * 32;
    const int l15 = lane & 15, lhi = lane >> 4;
    const int koff8 = lhi * 8, koff16 = lhi * 16;

    int arow[2], ncol[2];
    arow[0] = m0 + l15;       arow[1] = m0 + 16 + l15;
    ncol[0] = n0 + l15;       ncol[1] = n0 + 16 + l15;

    v8f acc[2][2];
#pragma unroll
    for (int ni = 0; ni < 2; ni++) {
        float bv = bias[ncol[ni]];
#pragma unroll
        for (int mi = 0; mi < 2; mi++)
#pragma unroll
            for (int r = 0; r < 8; r++) acc[mi][ni][r] = bv;
    }
    float g[2] = {1.0f, 1.0f};
    if constexpr (AMODE == 1) { g[0] = gains[arow[0]]; g[1] = gains[arow[1]]; }

    for (int k0 = 0; k0 < K; k0 += 32) {
        v16bf a[2], b[2];
#pragma unroll
        for (int mi = 0; mi < 2; mi++) {
            if constexpr (AMODE == 2) {
                const bf16_t* Ap = (const bf16_t*)Av + (size_t)arow[mi] * K + k0;
                if (k0 + 32 < K) __builtin_prefetch(Ap + 32, 0, 3);
#pragma unroll
                for (int i = 0; i < 8; i++) {
                    a[mi][i] = Ap[koff16 + i];
                    a[mi][8 + i] = Ap[koff16 + 8 + i];
                }
            } else {
                const float* Ap = (const float*)Av + (size_t)arow[mi] * K + k0;
                if (k0 + 32 < K) __builtin_prefetch(Ap + 32, 0, 3);
#pragma unroll
                for (int i = 0; i < 8; i++) {
                    a[mi][i]     = (bf16_t)(Ap[koff16 + i] * g[mi]);
                    a[mi][8 + i] = (bf16_t)(Ap[koff16 + 8 + i] * g[mi]);
                }
            }
        }
#pragma unroll
        for (int ni = 0; ni < 2; ni++) {
            const bf16_t* Bp = Bt + (size_t)ncol[ni] * K + k0 + koff16;
            if (k0 + 32 < K) __builtin_prefetch(Bp + 32, 0, 3);
#pragma unroll
            for (int i = 0; i < 16; i++) b[ni][i] = Bp[i];
        }
#pragma unroll
        for (int mi = 0; mi < 2; mi++)
#pragma unroll
            for (int ni = 0; ni < 2; ni++)
                acc[mi][ni] = __builtin_amdgcn_wmma_f32_16x16x32_bf16(
                    false, a[mi], false, b[ni], (short)0, acc[mi][ni], false, false);
    }

#pragma unroll
    for (int mi = 0; mi < 2; mi++) {
        const int orow0 = m0 + mi * 16 + lhi * 8;
        if constexpr (EPI == 0) {
            float* O = (float*)Out;
#pragma unroll
            for (int r = 0; r < 8; r++) {
                size_t rb = (size_t)(orow0 + r) * N;
                O[rb + ncol[0]] = acc[mi][0][r];
                O[rb + ncol[1]] = acc[mi][1][r];
            }
        } else {
            bf16_t* O = (bf16_t*)Out;
#pragma unroll
            for (int r = 0; r < 8; r++) {
                int orow = orow0 + r;
                float gg = gains[orow];
                size_t rb = (size_t)orow * N;
                float s0 = 1.0f / (1.0f + expf(-acc[mi][0][r]));
                float s1 = 1.0f / (1.0f + expf(-acc[mi][1][r]));
                O[rb + ncol[0]] = (bf16_t)(s0 * gg);
                O[rb + ncol[1]] = (bf16_t)(s1 * gg);
            }
        }
    }
}

// ---------------------------------------------------------------- GIF membrane scan
// chains = B*Hc, scan over S with row stride Hc; software-pipelined load of the
// next timestep so the global load overlaps the expf of the current step.
template <bool OUT_BF16>
__global__ void k_scan(const float* cur, void* out, int chains, int S, int Hc) {
    int idx = blockIdx.x * blockDim.x + threadIdx.x;
    if (idx >= chains) return;
    int b = idx / Hc, h = idx - b * Hc;
    size_t base = (size_t)b * S * Hc + h;
    float v = 0.0f;
    float c = cur[base];
    for (int s = 0; s < S; s++) {
        size_t p = base + (size_t)s * Hc;
        float cnext = (s + 1 < S) ? cur[p + Hc] : 0.0f;
        v = 0.9f * v + c;
        float sg = 1.0f / (1.0f + expf(-4.0f * (v - 1.0f)));
        v -= sg;  // thr = 1.0
        if constexpr (OUT_BF16) ((bf16_t*)out)[p] = (bf16_t)sg;
        else                    ((float*)out)[p]  = sg;
        c = cnext;
    }
}

// ---------------------------------------------------------------- router + top-2 combine
__global__ void k_router(const float* __restrict__ cont, const float* __restrict__ rW1,
                         const float* __restrict__ rb1, const float* __restrict__ rW2,
                         const float* __restrict__ rb2, const float* __restrict__ gains,
                         float* __restrict__ combine, int Ntok) {
    __shared__ float sW1[64 * 64];
    __shared__ float sW2[64 * 8];
    __shared__ float sb1[64];
    __shared__ float sb2[8];
    for (int i = threadIdx.x; i < 64 * 64; i += blockDim.x) sW1[i] = rW1[i];
    for (int i = threadIdx.x; i < 64 * 8; i += blockDim.x) sW2[i] = rW2[i];
    if (threadIdx.x < 64) sb1[threadIdx.x] = rb1[threadIdx.x];
    if (threadIdx.x < 8)  sb2[threadIdx.x] = rb2[threadIdx.x];
    __syncthreads();
    int n = blockIdx.x * blockDim.x + threadIdx.x;
    if (n >= Ntok) return;

    float c[64];
    const float* cr = cont + (size_t)n * 64;
#pragma unroll
    for (int k = 0; k < 64; k++) c[k] = cr[k];
    float t[64];
    for (int j = 0; j < 64; j++) {
        float s = sb1[j];
#pragma unroll
        for (int k = 0; k < 64; k++) s += c[k] * sW1[k * 64 + j];
        t[j] = tanhf(s);
    }
    float g = gains[n];
    float lg[8];
#pragma unroll
    for (int e = 0; e < 8; e++) {
        float s = sb2[e];
        for (int j = 0; j < 64; j++) s += t[j] * sW2[j * 8 + e];
        lg[e] = s * g;
    }
    float mx = lg[0];
#pragma unroll
    for (int e = 1; e < 8; e++) mx = fmaxf(mx, lg[e]);
    float p[8];
#pragma unroll
    for (int e = 0; e < 8; e++) p[e] = expf(lg[e] - mx);
    int i1 = 0; float pv1 = p[0];
#pragma unroll
    for (int e = 1; e < 8; e++) if (p[e] > pv1) { pv1 = p[e]; i1 = e; }
    int i2 = -1; float pv2 = -1.0f;
#pragma unroll
    for (int e = 0; e < 8; e++) if (e != i1 && p[e] > pv2) { pv2 = p[e]; i2 = e; }
    float inv = 1.0f / (pv1 + pv2);   // softmax denom cancels in top-k renorm
    float* cb = combine + (size_t)n * 8;
#pragma unroll
    for (int e = 0; e < 8; e++) cb[e] = 0.0f;
    cb[i1] = pv1 * inv;
    cb[i2] = pv2 * inv;
}

// ---------------------------------------------------------------- fused 8-expert MLP
// moe[n,0:64] = sum_e combine[n,e] * ( relu(cont @ W1[e] + b1[e]) @ W2[e] + b2[e] )
// One wave per 16-token tile; hidden [16 x 32] chunk round-trips through LDS to
// convert WMMA C-layout -> A-layout. Avoids the 2x134MB [E,N,H/2] intermediate.
__global__ void k_experts(const float* __restrict__ cont, const bf16_t* __restrict__ W1t,
                          const float* __restrict__ b1, const bf16_t* __restrict__ W2t,
                          const float* __restrict__ b2, const float* __restrict__ combine,
                          float* __restrict__ moe, int Ntok) {
    __shared__ bf16_t hbuf_all[8][16 * 32];
    const int lane = threadIdx.x & 31;
    const int wib = threadIdx.x >> 5;
    const int wave = (int)(((size_t)blockIdx.x * blockDim.x + threadIdx.x) >> 5);
    const int m0 = wave * 16;
    if (m0 >= Ntok) return;
    bf16_t* hbuf = hbuf_all[wib];
    const int l15 = lane & 15, lhi = lane >> 4;
    const int koff8 = lhi * 8, koff16 = lhi * 16;

    // cont A-fragments for full K=64 (two k-steps), f32 -> bf16 on the fly
    v16bf a0, a1;
    {
        const float* cr = cont + (size_t)(m0 + l15) * 64;
#pragma unroll
        for (int i = 0; i < 8; i++) {
            a0[i]     = (bf16_t)cr[koff16 + i];
            a0[8 + i] = (bf16_t)cr[koff16 + 8 + i];
            a1[i]     = (bf16_t)cr[32 + koff16 + i];
            a1[8 + i] = (bf16_t)cr[32 + koff16 + 8 + i];
        }
    }
    v8f macc[4];
#pragma unroll
    for (int nt = 0; nt < 4; nt++)
#pragma unroll
        for (int r = 0; r < 8; r++) macc[nt][r] = 0.0f;

    for (int e = 0; e < NEXP; e++) {
        float s[8];
#pragma unroll
        for (int r = 0; r < 8; r++) s[r] = combine[(size_t)(m0 + lhi * 8 + r) * 8 + e];
        v8f eacc[4];
#pragma unroll
        for (int nt = 0; nt < 4; nt++) {
            float bv = b2[e * 64 + nt * 16 + l15];
#pragma unroll
            for (int r = 0; r < 8; r++) eacc[nt][r] = bv;
        }
        for (int kk = 0; kk < HHALF; kk += 32) {
            // h chunk: two 16x16 tiles over hidden cols [kk, kk+32)
#pragma unroll
            for (int ct = 0; ct < 2; ct++) {
                int col = kk + ct * 16 + l15;
                float bv = b1[e * HHALF + col];
                v8f h;
#pragma unroll
                for (int r = 0; r < 8; r++) h[r] = bv;
                const bf16_t* Bp = W1t + ((size_t)e * HHALF + col) * 64 + koff16;
                v16bf bb;
#pragma unroll
                for (int i = 0; i < 16; i++) bb[i] = Bp[i];
                h = __builtin_amdgcn_wmma_f32_16x16x32_bf16(false, a0, false, bb, (short)0, h,
                                                            false, false);
#pragma unroll
                for (int i = 0; i < 16; i++) bb[i] = Bp[32 + i];
                h = __builtin_amdgcn_wmma_f32_16x16x32_bf16(false, a1, false, bb, (short)0, h,
                                                            false, false);
                // relu + stage to LDS row-major [16 tokens][32 cols]
#pragma unroll
                for (int r = 0; r < 8; r++) {
                    float v = fmaxf(h[r], 0.0f);
                    hbuf[(lhi * 8 + r) * 32 + ct * 16 + l15] = (bf16_t)v;
                }
            }
            // same-wave DS ordering: wait stores, then gather A-layout fragment
            asm volatile("s_wait_dscnt 0" ::: "memory");
            v16bf ah;
            const bf16_t* hp = hbuf + l15 * 32;
#pragma unroll
            for (int i = 0; i < 8; i++) { ah[i] = hp[koff16 + i]; ah[8 + i] = hp[koff16 + 8 + i]; }
#pragma unroll
            for (int nt = 0; nt < 4; nt++) {
                const bf16_t* B2p = W2t + ((size_t)e * 64 + nt * 16 + l15) * HHALF + kk + koff16;
                v16bf b2f;
#pragma unroll
                for (int i = 0; i < 16; i++) b2f[i] = B2p[i];
                eacc[nt] = __builtin_amdgcn_wmma_f32_16x16x32_bf16(false, ah, false, b2f,
                                                                   (short)0, eacc[nt], false, false);
            }
        }
#pragma unroll
        for (int nt = 0; nt < 4; nt++)
#pragma unroll
            for (int r = 0; r < 8; r++) macc[nt][r] += s[r] * eacc[nt][r];
    }
#pragma unroll
    for (int nt = 0; nt < 4; nt++)
#pragma unroll
        for (int r = 0; r < 8; r++)
            moe[(size_t)(m0 + lhi * 8 + r) * 64 + nt * 16 + l15] = macc[nt][r];
}

// ---------------------------------------------------------------- layernorm (in place)
__global__ void k_ln(float* __restrict__ x, const float* __restrict__ g,
                     const float* __restrict__ b, int Dd) {
    __shared__ float ssum[8], ssq[8];
    float* p = x + (size_t)blockIdx.x * Dd;
    float sum = 0.0f, sq = 0.0f;
    for (int i = threadIdx.x; i < Dd; i += blockDim.x) { float v = p[i]; sum += v; sq += v * v; }
    for (int o = 16; o > 0; o >>= 1) { sum += __shfl_xor(sum, o); sq += __shfl_xor(sq, o); }
    int wid = threadIdx.x >> 5, lane = threadIdx.x & 31;
    if (lane == 0) { ssum[wid] = sum; ssq[wid] = sq; }
    __syncthreads();
    if (threadIdx.x == 0) {
        float ts = 0, tq = 0;
        for (int w = 0; w < (int)(blockDim.x >> 5); w++) { ts += ssum[w]; tq += ssq[w]; }
        ssum[0] = ts; ssq[0] = tq;
    }
    __syncthreads();
    float mu = ssum[0] / Dd;
    float var = ssq[0] / Dd - mu * mu;
    float rstd = rsqrtf(var + 1e-5f);
    for (int i = threadIdx.x; i < Dd; i += blockDim.x)
        p[i] = (p[i] - mu) * rstd * g[i] + b[i];
}

// ================================================================ launch
extern "C" void kernel_launch(void* const* d_in, const int* in_sizes, int n_in,
                              void* d_out, int out_size, void* d_ws, size_t ws_size,
                              hipStream_t stream) {
    const float* x     = (const float*)d_in[0];
    const int*   ids   = (const int*)d_in[1];
    const float* ptab  = (const float*)d_in[2];
    const float* enc_W = (const float*)d_in[3];
    const float* enc_b = (const float*)d_in[4];
    const float* s2c_W = (const float*)d_in[5];
    const float* s2c_b = (const float*)d_in[6];
    const float* r_W1  = (const float*)d_in[7];
    const float* r_b1  = (const float*)d_in[8];
    const float* r_W2  = (const float*)d_in[9];
    const float* r_b2  = (const float*)d_in[10];
    const float* e_W1  = (const float*)d_in[11];
    const float* e_b1  = (const float*)d_in[12];
    const float* e_W2  = (const float*)d_in[13];
    const float* e_b2  = (const float*)d_in[14];
    const float* c2s_W = (const float*)d_in[15];
    const float* c2s_b = (const float*)d_in[16];
    const float* dec_W = (const float*)d_in[17];
    const float* dec_b = (const float*)d_in[18];
    const float* ln_g  = (const float*)d_in[19];
    const float* ln_b  = (const float*)d_in[20];
    float* out = (float*)d_out;

    uint8_t* ws = (uint8_t*)d_ws;
    size_t off = 0;
    auto alloc = [&](size_t bytes) -> void* {
        void* p = ws + off;
        off += (bytes + 255) & ~(size_t)255;
        return p;
    };
    float*  gains   = (float*)alloc((size_t)NTOK * 4);
    bf16_t* enc_Wt  = (bf16_t*)alloc((size_t)HDIM * DDIM * 2);   // [2048][1024]
    bf16_t* s2c_Wt  = (bf16_t*)alloc((size_t)MMOE * HDIM * 2);   // [64][2048]
    bf16_t* eW1t    = (bf16_t*)alloc((size_t)NEXP * HHALF * MMOE * 2); // [8][1024][64]
    bf16_t* eW2t    = (bf16_t*)alloc((size_t)NEXP * MMOE * HHALF * 2); // [8][64][1024]
    bf16_t* c2s_Wt  = (bf16_t*)alloc((size_t)HDIM * MMOE * 2);   // [2048][64]
    bf16_t* dec_Wt  = (bf16_t*)alloc((size_t)DDIM * HDIM * 2);   // [1024][2048]
    float*  cur1    = (float*)alloc((size_t)NTOK * HDIM * 4);
    bf16_t* spikes  = (bf16_t*)alloc((size_t)NTOK * HDIM * 2);
    float*  cont    = (float*)alloc((size_t)NTOK * MMOE * 4);
    float*  combine = (float*)alloc((size_t)NTOK * NEXP * 4);
    float*  moe     = (float*)alloc((size_t)NTOK * MMOE * 4);
    bf16_t* sp_avg  = (bf16_t*)alloc((size_t)NTOK * HDIM * 2);

    k_gains<<<NTOK / 256, 256, 0, stream>>>(ids, ptab, gains, NTOK);

    k_transpose_bf16<<<1024, 256, 0, stream>>>(enc_W, enc_Wt, 1, DDIM, HDIM);
    k_transpose_bf16<<<512, 256, 0, stream>>>(s2c_W, s2c_Wt, 1, HDIM, MMOE);
    k_transpose_bf16<<<512, 256, 0, stream>>>(e_W1, eW1t, NEXP, MMOE, HHALF);
    k_transpose_bf16<<<512, 256, 0, stream>>>(e_W2, eW2t, NEXP, HHALF, MMOE);
    k_transpose_bf16<<<256, 256, 0, stream>>>(c2s_W, c2s_Wt, 1, MMOE, HDIM);
    k_transpose_bf16<<<1024, 256, 0, stream>>>(dec_W, dec_Wt, 1, HDIM, DDIM);

    // G1: cur1 = (x * gain) @ enc_W + enc_b        [8192,2048], K=1024
    { int waves = (NTOK / 32) * (HDIM / 32);
      k_gemm_wmma<1, 0><<<waves * 32 / 256, 256, 0, stream>>>(x, enc_Wt, enc_b, gains, cur1,
                                                              NTOK, HDIM, DDIM); }
    // GIF scan #1 -> spikes bf16
    k_scan<true><<<(NB * HDIM) / 256, 256, 0, stream>>>(cur1, spikes, NB * HDIM, SSEQ, HDIM);
    // G2: cont = spikes @ s2c_W + b                [8192,64], K=2048
    { int waves = (NTOK / 32) * (MMOE / 32);
      k_gemm_wmma<2, 0><<<waves * 32 / 256, 256, 0, stream>>>(spikes, s2c_Wt, s2c_b, gains,
                                                              cont, NTOK, MMOE, HDIM); }
    // router -> combine [8192,8]
    k_router<<<NTOK / 256, 256, 0, stream>>>(cont, r_W1, r_b1, r_W2, r_b2, gains, combine, NTOK);
    // fused experts -> moe [8192,64]
    k_experts<<<(NTOK / 16) / 8, 256, 0, stream>>>(cont, eW1t, e_b1, eW2t, e_b2, combine, moe,
                                                   NTOK);
    // G4: sp_avg = bf16( sigmoid(moe @ c2s_W + b) * gain )   [8192,2048], K=64
    { int waves = (NTOK / 32) * (HDIM / 32);
      k_gemm_wmma<0, 1><<<waves * 32 / 256, 256, 0, stream>>>(moe, c2s_Wt, c2s_b, gains,
                                                              sp_avg, NTOK, HDIM, MMOE); }
    // G5: cur2 = sp_avg @ dec_W + dec_b  -> written directly into d_out  [8192,1024], K=2048
    { int waves = (NTOK / 32) * (DDIM / 32);
      k_gemm_wmma<2, 0><<<waves * 32 / 256, 256, 0, stream>>>(sp_avg, dec_Wt, dec_b, gains,
                                                              out, NTOK, DDIM, HDIM); }
    // GIF scan #2 in place on d_out
    k_scan<false><<<(NB * DDIM) / 256, 256, 0, stream>>>(out, out, NB * DDIM, SSEQ, DDIM);
    // layernorm in place on d_out
    k_ln<<<NTOK, 256, 0, stream>>>(out, ln_g, ln_b, DDIM);

    (void)in_sizes; (void)n_in; (void)out_size; (void)ws_size;
}
